// NMT_16045997818573
// MI455X (gfx1250) — matrix operand order repaired and checked by reading, not compiled, not run on hardware
//
#include <hip/hip_runtime.h>
#include <hip/hip_bf16.h>

// ---------------------------------------------------------------------------
// NMT seq2seq forward for MI455X (gfx1250, wave32, WMMA).
//
//   1. gather embeddings (bf16)
//   2. shuffle weights into WMMA B-fragment tiled bf16 layout (one-time)
//   3. hoisted input projections PF/PR/PD (bias folded)
//   4. persistent bidirectional-LSTM encoder (grid=2, 16 waves/WGP)
//   5. persistent decoder+attention (grid=1, 32 waves)
//   6. generator GEMM (bf16 WMMA, f32 accum) -> logits in d_out
//   7. row-wise log_softmax in place
//
// B-tiled layout: for W (N x K):  dst[((nt*(K/32)+kt)*32 + lane)*16 + e]
//   = W[nt*16+lane%16][kt*32 + 16*(e/8) + 8*(lane/16) + e%8]
// so a lane's whole 16-element fragment is 32 contiguous bytes.
// ---------------------------------------------------------------------------

typedef __bf16  bf16;
typedef __attribute__((ext_vector_type(16))) __bf16 v16bf;
typedef __attribute__((ext_vector_type(8)))  __bf16 v8bf;
typedef __attribute__((ext_vector_type(8)))  float  v8f;

#define S_LEN 64
#define T_LEN 50
#define BATCH 32
#define EMB   512
#define HH    512          // H
#define DH    1024         // decoder hidden = 2H
#define VOCAB 32000
#define DROWS 1568         // (T-1)*B

__device__ __forceinline__ v8f wmma_bf16(v16bf a, v16bf b, v8f c) {
  return __builtin_amdgcn_wmma_f32_16x16x32_bf16(false, a, false, b, (short)0, c,
                                                 false, false);
}

__device__ __forceinline__ v16bf combine16(v8bf lo, v8bf hi) {
  return __builtin_shufflevector(lo, hi, 0, 1, 2, 3, 4, 5, 6, 7,
                                 8, 9, 10, 11, 12, 13, 14, 15);
}

// A fragment: 16x32 bf16 from row-major source (lda bf16 elements).
// lane<16: row m=lane, K={0..7,16..23}; lane>=16: same rows, K={8..15,24..31}.
// Two 16B vector loads per lane (global_load_b128 / ds_load_b128).
__device__ __forceinline__ v16bf load_a_frag(const bf16* A, int lda, int mbase,
                                             int kb, int lane) {
  const int m = mbase + (lane & 15);
  const bf16* p = A + (size_t)m * lda + kb + ((lane >> 4) << 3);
  v8bf lo = *(const v8bf*)p;
  v8bf hi = *(const v8bf*)(p + 16);
  return combine16(lo, hi);
}

// B fragment from pre-tiled weights: one lane = 32 contiguous bytes.
__device__ __forceinline__ v16bf load_b_tiled(const bf16* Wt, int ktiles, int nt,
                                              int kt, int lane) {
  const bf16* p = Wt + ((((size_t)nt * ktiles + kt) << 5) + lane) * 16;
  v8bf lo = *(const v8bf*)p;
  v8bf hi = *(const v8bf*)(p + 8);
  return combine16(lo, hi);
}

__device__ __forceinline__ float sigm(float x) { return 1.0f / (1.0f + __expf(-x)); }

#define ZERO_V8F {0.f,0.f,0.f,0.f,0.f,0.f,0.f,0.f}

// ---------------------------------------------------------------------------
// Elementwise helpers
// ---------------------------------------------------------------------------
__global__ void gather_k(const float* __restrict__ emb, const int* __restrict__ idx,
                         bf16* __restrict__ dst, int rows) {
  int i = blockIdx.x * blockDim.x + threadIdx.x;
  if (i < rows * EMB) {
    int r = i >> 9, c = i & 511;
    dst[i] = (bf16)emb[(size_t)idx[r] * EMB + c];
  }
}

// fp32 (N x K, row-major, leading dim src_ld, col offset src_c0) -> B-tiled bf16
__global__ void cvt_tileB_k(const float* __restrict__ src, int src_ld, int src_c0,
                            bf16* __restrict__ dst, int N, int K) {
  int i = blockIdx.x * blockDim.x + threadIdx.x;
  if (i >= N * K) return;
  int e = i & 15, lane = (i >> 4) & 31, tile = i >> 9;
  int ktiles = K >> 5;
  int kt = tile % ktiles, nt = tile / ktiles;
  dst[i] = (bf16)src[(size_t)(nt * 16 + (lane & 15)) * src_ld + src_c0 + kt * 32 + ((e >> 3) << 4) + ((lane >> 4) << 3) + (e & 7)];
}

// Wcat = [dec_w_ih[:,512:1536] | dec_w_hh]  (N=4096, K=2048) -> B-tiled bf16
__global__ void cvt_tile_wcat_k(const float* __restrict__ wih,
                                const float* __restrict__ whh,
                                bf16* __restrict__ dst) {
  int i = blockIdx.x * blockDim.x + threadIdx.x;
  if (i >= 4 * DH * 2048) return;
  int e = i & 15, lane = (i >> 4) & 31, tile = i >> 9;
  int kt = tile & 63, nt = tile >> 6;  // ktiles = 64
  int n = nt * 16 + (lane & 15), k = kt * 32 + ((e >> 3) << 4) + ((lane >> 4) << 3) + (e & 7);
  float v = (k < DH) ? wih[(size_t)n * 1536 + EMB + k]
                     : whh[(size_t)n * DH + (k - DH)];
  dst[i] = (bf16)v;
}

// ---------------------------------------------------------------------------
// Generic WMMA GEMM: C(M,N) = A(M,K) @ W^T + bias, W pre-tiled.
// Each wave: 1 Mtile x 4 Ntiles.
// ---------------------------------------------------------------------------
__global__ void gemm_bf16_k(const bf16* __restrict__ A, int lda,
                            const bf16* __restrict__ Bt,
                            const float* __restrict__ bias,
                            float* __restrict__ C, int ldc,
                            int mtiles, int ngroups, int K) {
  const int lane = threadIdx.x & 31;
  const int wave = (blockIdx.x * blockDim.x + threadIdx.x) >> 5;
  if (wave >= mtiles * ngroups) return;
  const int mt = wave / ngroups, ng = wave - mt * ngroups;
  const int ktiles = K >> 5;

  v8f acc[4] = {ZERO_V8F, ZERO_V8F, ZERO_V8F, ZERO_V8F};
#pragma unroll 1
  for (int kt = 0; kt < ktiles; ++kt) {
    if (kt + 1 < ktiles)  // pull next contiguous B tile toward the WGP
      __builtin_prefetch(Bt + ((((size_t)(ng * 4) * ktiles + kt + 1) << 5) + lane) * 16, 0, 3);
    v16bf a = load_a_frag(A, lda, mt * 16, kt * 32, lane);
#pragma unroll
    for (int s = 0; s < 4; ++s) {
      v16bf b = load_b_tiled(Bt, ktiles, ng * 4 + s, kt, lane);
      acc[s] = wmma_bf16(a, b, acc[s]);
    }
  }
#pragma unroll
  for (int s = 0; s < 4; ++s) {
#pragma unroll
    for (int r = 0; r < 8; ++r) {
      int row = mt * 16 + r + ((lane >> 4) << 3);
      int col = (ng * 4 + s) * 16 + (lane & 15);
      float v = acc[s][r] + (bias ? bias[col] : 0.0f);
      C[(size_t)row * ldc + col] = v;
    }
  }
}

// ---------------------------------------------------------------------------
// Persistent bidirectional encoder LSTM.  grid=2 (dir), block=512 (16 waves).
// gates(32,2048) = h @ W_hh^T + P[tt]; all 4 gates of an (m,j) element land in
// the same lane -> in-register LSTM cell.  LDS: h double-buffered bf16 + c f32.
// ---------------------------------------------------------------------------
__global__ void encoder_k(const float* __restrict__ PF, const float* __restrict__ PR,
                          const bf16* __restrict__ whh_f, const bf16* __restrict__ whh_r,
                          float* __restrict__ ctx, float* __restrict__ hfin,
                          float* __restrict__ cfin) {
  extern __shared__ char smem[];
  bf16* h_s = (bf16*)smem;                                      // [2][32*512]
  float* c_s = (float*)(smem + 2 * BATCH * HH * sizeof(bf16));  // [32*512]

  const int dir = blockIdx.x;
  const float* P = dir ? PR : PF;
  const bf16* W = dir ? whh_r : whh_f;  // tiled, ktiles = 16
  const int tid = threadIdx.x, lane = tid & 31, wave = tid >> 5;

  for (int i = tid; i < 2 * BATCH * HH; i += blockDim.x) h_s[i] = (bf16)0.0f;
  for (int i = tid; i < BATCH * HH; i += blockDim.x) c_s[i] = 0.0f;
  __syncthreads();

#pragma unroll 1
  for (int t = 0; t < S_LEN; ++t) {
    const int tt = dir ? (S_LEN - 1 - t) : t;
    const bf16* hcur = h_s + (t & 1) * (BATCH * HH);
    bf16* hnxt = h_s + ((t + 1) & 1) * (BATCH * HH);

#pragma unroll 1
    for (int pass = 0; pass < 2; ++pass) {
      const int jt = wave + 16 * pass;  // 0..31 (j-tiles of 512)
      v8f acc[2][4] = {{ZERO_V8F, ZERO_V8F, ZERO_V8F, ZERO_V8F},
                       {ZERO_V8F, ZERO_V8F, ZERO_V8F, ZERO_V8F}};
#pragma unroll 1
      for (int kt = 0; kt < 16; ++kt) {
        v16bf a0 = load_a_frag(hcur, HH, 0, kt * 32, lane);
        v16bf a1 = load_a_frag(hcur, HH, 16, kt * 32, lane);
#pragma unroll
        for (int g = 0; g < 4; ++g) {
          v16bf b = load_b_tiled(W, 16, g * 32 + jt, kt, lane);
          acc[0][g] = wmma_bf16(a0, b, acc[0][g]);
          acc[1][g] = wmma_bf16(a1, b, acc[1][g]);
        }
      }
      const int j = jt * 16 + (lane & 15);
#pragma unroll
      for (int mt = 0; mt < 2; ++mt) {
#pragma unroll
        for (int r = 0; r < 8; ++r) {
          const int m = mt * 16 + r + ((lane >> 4) << 3);
          const float* prow = P + ((size_t)tt * BATCH + m) * (4 * HH);
          float gi = acc[mt][0][r] + prow[j];
          float gf = acc[mt][1][r] + prow[HH + j];
          float gg = acc[mt][2][r] + prow[2 * HH + j];
          float go = acc[mt][3][r] + prow[3 * HH + j];
          float cold = c_s[m * HH + j];
          float cn = sigm(gf) * cold + sigm(gi) * tanhf(gg);
          float hn = sigm(go) * tanhf(cn);
          c_s[m * HH + j] = cn;
          hnxt[m * HH + j] = (bf16)hn;
          ctx[((size_t)tt * BATCH + m) * DH + dir * HH + j] = hn;
          if (t == S_LEN - 1) {
            hfin[(size_t)dir * BATCH * HH + m * HH + j] = hn;
            cfin[(size_t)dir * BATCH * HH + m * HH + j] = cn;
          }
        }
      }
    }
    __syncthreads();
  }
}

// ---------------------------------------------------------------------------
// Persistent decoder + Luong attention.  grid=1, block=1024 (32 waves).
// U (LDS bf16 32x2048) holds [out | h] feeding gates GEMM against tiled Wcat.
// ---------------------------------------------------------------------------
__global__ void decoder_k(const float* __restrict__ PD, const bf16* __restrict__ wcat,
                          const bf16* __restrict__ w_ain, const bf16* __restrict__ w_aout,
                          const float* __restrict__ ctx, const int* __restrict__ src_mask,
                          const float* __restrict__ hf, const float* __restrict__ hr,
                          const float* __restrict__ cf, const float* __restrict__ cr,
                          float* __restrict__ c_glob, bf16* __restrict__ dec_out) {
  extern __shared__ char smem[];
  bf16* U = (bf16*)smem;                                  // 32 x 2048  (128KB)
  bf16* cvecb = (bf16*)(smem + BATCH * 2048 * 2);         // 32 x 1024  (64KB) (+ h stage)
  bf16* Qb = (bf16*)(smem + BATCH * 2048 * 2 + BATCH * DH * 2);        // 32 x 1024
  float* a_s = (float*)(smem + BATCH * 2048 * 2 + 2 * BATCH * DH * 2); // 32 x 64

  const int tid = threadIdx.x, lane = tid & 31, wave = tid >> 5;

  for (int i = tid; i < BATCH * DH; i += 1024) {
    int b = i >> 10, d = i & 1023;
    float hv = (d < HH) ? hf[b * HH + d] : hr[b * HH + d - HH];
    float cv = (d < HH) ? cf[b * HH + d] : cr[b * HH + d - HH];
    U[b * 2048 + d] = (bf16)hv;       // out slot
    U[b * 2048 + DH + d] = (bf16)hv;  // h slot
    c_glob[i] = cv;
  }
  __syncthreads();

#pragma unroll 1
  for (int t = 0; t < T_LEN - 1; ++t) {
    // ---- phase 1: gates GEMM (M=32,N=4096,K=2048) + LSTM cell (2 reg passes)
#pragma unroll 1
    for (int pass = 0; pass < 2; ++pass) {
      const int jt = wave + 32 * pass;  // 0..63 (j-tiles of 1024)
      v8f acc[2][4] = {{ZERO_V8F, ZERO_V8F, ZERO_V8F, ZERO_V8F},
                       {ZERO_V8F, ZERO_V8F, ZERO_V8F, ZERO_V8F}};
#pragma unroll 1
      for (int kt = 0; kt < 64; ++kt) {
        v16bf a0 = load_a_frag(U, 2048, 0, kt * 32, lane);
        v16bf a1 = load_a_frag(U, 2048, 16, kt * 32, lane);
#pragma unroll
        for (int g = 0; g < 4; ++g) {
          v16bf b = load_b_tiled(wcat, 64, g * 64 + jt, kt, lane);
          acc[0][g] = wmma_bf16(a0, b, acc[0][g]);
          acc[1][g] = wmma_bf16(a1, b, acc[1][g]);
        }
      }
      const int j = jt * 16 + (lane & 15);
#pragma unroll
      for (int mt = 0; mt < 2; ++mt) {
#pragma unroll
        for (int r = 0; r < 8; ++r) {
          const int m = mt * 16 + r + ((lane >> 4) << 3);
          const float* prow = PD + ((size_t)t * BATCH + m) * (4 * DH);
          float gi = acc[mt][0][r] + prow[j];
          float gf = acc[mt][1][r] + prow[DH + j];
          float gg = acc[mt][2][r] + prow[2 * DH + j];
          float go = acc[mt][3][r] + prow[3 * DH + j];
          float cold = c_glob[m * DH + j];
          float cn = sigm(gf) * cold + sigm(gi) * tanhf(gg);
          float hn = sigm(go) * tanhf(cn);
          c_glob[m * DH + j] = cn;
          cvecb[m * DH + j] = (bf16)hn;  // stage new h (U still being read)
        }
      }
    }
    __syncthreads();
    // ---- phase 2: commit staged h into U.h
    for (int i = tid; i < BATCH * DH; i += 1024) {
      int b = i >> 10, d = i & 1023;
      U[b * 2048 + DH + d] = cvecb[i];
    }
    __syncthreads();

    // ---- phase 3: q = h @ attn_in^T  (M=32,N=1024,K=1024)
    {
      v8f q[2][2] = {{ZERO_V8F, ZERO_V8F}, {ZERO_V8F, ZERO_V8F}};
#pragma unroll 1
      for (int kt = 0; kt < 32; ++kt) {
        v16bf a0 = load_a_frag(U + DH, 2048, 0, kt * 32, lane);
        v16bf a1 = load_a_frag(U + DH, 2048, 16, kt * 32, lane);
#pragma unroll
        for (int s = 0; s < 2; ++s) {
          v16bf b = load_b_tiled(w_ain, 32, wave * 2 + s, kt, lane);
          q[0][s] = wmma_bf16(a0, b, q[0][s]);
          q[1][s] = wmma_bf16(a1, b, q[1][s]);
        }
      }
#pragma unroll
      for (int mt = 0; mt < 2; ++mt)
#pragma unroll
        for (int s = 0; s < 2; ++s)
#pragma unroll
          for (int r = 0; r < 8; ++r) {
            int m = mt * 16 + r + ((lane >> 4) << 3);
            int n = (wave * 2 + s) * 16 + (lane & 15);
            Qb[m * DH + n] = (bf16)q[mt][s][r];
          }
    }
    __syncthreads();

    // ---- phase 4a: align scores (masked)
    for (int p = tid; p < BATCH * S_LEN; p += 1024) {
      int b = p >> 6, s = p & 63;
      const float* cp = ctx + ((size_t)s * BATCH + b) * DH;
      float d = 0.0f;
      for (int k = 0; k < DH; ++k) d += (float)Qb[b * DH + k] * cp[k];
      a_s[b * S_LEN + s] = src_mask[s * BATCH + b] ? d : -1.0e6f;
    }
    __syncthreads();
    // ---- phase 4b: softmax over S (one wave per batch row, wave32)
    {
      const int b = wave;
      float v0 = a_s[b * S_LEN + lane], v1 = a_s[b * S_LEN + 32 + lane];
      float mx = fmaxf(v0, v1);
#pragma unroll
      for (int off = 16; off > 0; off >>= 1) mx = fmaxf(mx, __shfl_xor(mx, off));
      float e0 = __expf(v0 - mx), e1 = __expf(v1 - mx);
      float sm = e0 + e1;
#pragma unroll
      for (int off = 16; off > 0; off >>= 1) sm += __shfl_xor(sm, off);
      float inv = 1.0f / sm;
      a_s[b * S_LEN + lane] = e0 * inv;
      a_s[b * S_LEN + 32 + lane] = e1 * inv;
    }
    __syncthreads();
    // ---- phase 4c: context vector
    for (int i = tid; i < BATCH * DH; i += 1024) {
      int b = i >> 10, d = i & 1023;
      float acc = 0.0f;
      for (int s = 0; s < S_LEN; ++s)
        acc += a_s[b * S_LEN + s] * ctx[((size_t)s * BATCH + b) * DH + d];
      cvecb[i] = (bf16)acc;
    }
    __syncthreads();

    // ---- phase 5: attn_h = tanh([cvec, h] @ attn_out^T)  (M=32,N=1024,K=2048)
    {
      v8f o[2][2] = {{ZERO_V8F, ZERO_V8F}, {ZERO_V8F, ZERO_V8F}};
#pragma unroll 1
      for (int kt = 0; kt < 64; ++kt) {
        int kb = kt * 32;
        // piecewise A: k<1024 -> cvec ; k>=1024 -> U (lda 2048 lands in h slot)
        v16bf a0 = (kb < DH) ? load_a_frag(cvecb, DH, 0, kb, lane)
                             : load_a_frag(U, 2048, 0, kb, lane);
        v16bf a1 = (kb < DH) ? load_a_frag(cvecb, DH, 16, kb, lane)
                             : load_a_frag(U, 2048, 16, kb, lane);
#pragma unroll
        for (int s = 0; s < 2; ++s) {
          v16bf b = load_b_tiled(w_aout, 64, wave * 2 + s, kt, lane);
          o[0][s] = wmma_bf16(a0, b, o[0][s]);
          o[1][s] = wmma_bf16(a1, b, o[1][s]);
        }
      }
#pragma unroll
      for (int mt = 0; mt < 2; ++mt)
#pragma unroll
        for (int s = 0; s < 2; ++s)
#pragma unroll
          for (int r = 0; r < 8; ++r) {
            int m = mt * 16 + r + ((lane >> 4) << 3);
            int n = (wave * 2 + s) * 16 + (lane & 15);
            float hv = tanhf(o[mt][s][r]);
            U[m * 2048 + n] = (bf16)hv;  // new "out"
            dec_out[((size_t)t * BATCH + m) * DH + n] = (bf16)hv;
          }
    }
    __syncthreads();
  }
}

// ---------------------------------------------------------------------------
// In-place row log_softmax over V=32000.  grid = DROWS, block = 256.
// ---------------------------------------------------------------------------
__global__ void logsoftmax_k(float* __restrict__ out) {
  __shared__ float red[256];
  const int row = blockIdx.x, tid = threadIdx.x;
  float* p = out + (size_t)row * VOCAB;
  float mx = -3.4e38f;
  for (int c = tid; c < VOCAB; c += 256) mx = fmaxf(mx, p[c]);
  red[tid] = mx;
  __syncthreads();
  for (int s = 128; s > 0; s >>= 1) {
    if (tid < s) red[tid] = fmaxf(red[tid], red[tid + s]);
    __syncthreads();
  }
  mx = red[0];
  __syncthreads();
  float sm = 0.0f;
  for (int c = tid; c < VOCAB; c += 256) sm += __expf(p[c] - mx);
  red[tid] = sm;
  __syncthreads();
  for (int s = 128; s > 0; s >>= 1) {
    if (tid < s) red[tid] += red[tid + s];
    __syncthreads();
  }
  float lse = mx + __logf(red[0]);
  for (int c = tid; c < VOCAB; c += 256) p[c] = p[c] - lse;
}

// ---------------------------------------------------------------------------
extern "C" void kernel_launch(void* const* d_in, const int* in_sizes, int n_in,
                              void* d_out, int out_size, void* d_ws, size_t ws_size,
                              hipStream_t stream) {
  const int* src_idx = (const int*)d_in[0];
  const int* trg_idx = (const int*)d_in[1];
  const int* src_mask = (const int*)d_in[2];
  const float* emb_src = (const float*)d_in[4];
  const float* emb_trg = (const float*)d_in[5];
  const float* enc_w_ih_f = (const float*)d_in[6];
  const float* enc_w_hh_f = (const float*)d_in[7];
  const float* enc_b_f = (const float*)d_in[8];
  const float* enc_w_ih_r = (const float*)d_in[9];
  const float* enc_w_hh_r = (const float*)d_in[10];
  const float* enc_b_r = (const float*)d_in[11];
  const float* dec_w_ih = (const float*)d_in[12];
  const float* dec_w_hh = (const float*)d_in[13];
  const float* dec_b = (const float*)d_in[14];
  const float* attn_in = (const float*)d_in[15];
  const float* attn_out = (const float*)d_in[16];
  const float* gen_w = (const float*)d_in[17];
  const float* gen_b = (const float*)d_in[18];
  float* out = (float*)d_out;

  size_t o = 0;
  char* w = (char*)d_ws;
  auto alloc = [&](size_t bytes) { void* p = w + o; o += bytes; return p; };
  bf16* xsrc_bf   = (bf16*)alloc((size_t)S_LEN * BATCH * EMB * 2);
  bf16* xtrg_bf   = (bf16*)alloc((size_t)DROWS * EMB * 2);
  bf16* wih_f_bf  = (bf16*)alloc((size_t)4 * HH * EMB * 2);   // B-tiled
  bf16* wih_r_bf  = (bf16*)alloc((size_t)4 * HH * EMB * 2);   // B-tiled
  bf16* whh_f_bf  = (bf16*)alloc((size_t)4 * HH * HH * 2);    // B-tiled
  bf16* whh_r_bf  = (bf16*)alloc((size_t)4 * HH * HH * 2);    // B-tiled
  bf16* wih_e_bf  = (bf16*)alloc((size_t)4 * DH * EMB * 2);   // B-tiled
  bf16* wcat_bf   = (bf16*)alloc((size_t)4 * DH * 2048 * 2);  // B-tiled
  bf16* ain_bf    = (bf16*)alloc((size_t)DH * DH * 2);        // B-tiled
  bf16* aout_bf   = (bf16*)alloc((size_t)DH * 2048 * 2);      // B-tiled
  bf16* genw_bf   = (bf16*)alloc((size_t)VOCAB * DH * 2);     // B-tiled
  float* PF       = (float*)alloc((size_t)S_LEN * BATCH * 4 * HH * 4);
  float* PR       = (float*)alloc((size_t)S_LEN * BATCH * 4 * HH * 4);
  float* PD       = (float*)alloc((size_t)DROWS * 4 * DH * 4);
  float* ctx      = (float*)alloc((size_t)S_LEN * BATCH * DH * 4);
  float* hfin     = (float*)alloc((size_t)2 * BATCH * HH * 4);
  float* cfin     = (float*)alloc((size_t)2 * BATCH * HH * 4);
  float* c_glob   = (float*)alloc((size_t)BATCH * DH * 4);
  bf16* dec_out   = (bf16*)alloc((size_t)DROWS * DH * 2);
  (void)ws_size; (void)n_in; (void)in_sizes; (void)out_size;

  const int TB = 256;
  auto cdiv = [](int a, int b) { return (a + b - 1) / b; };

  // 1. embeddings
  gather_k<<<cdiv(S_LEN * BATCH * EMB, TB), TB, 0, stream>>>(emb_src, src_idx, xsrc_bf, S_LEN * BATCH);
  gather_k<<<cdiv(DROWS * EMB, TB), TB, 0, stream>>>(emb_trg, trg_idx, xtrg_bf, DROWS);

  // 2. weights -> WMMA B-tiled bf16
  cvt_tileB_k<<<cdiv(4 * HH * EMB, TB), TB, 0, stream>>>(enc_w_ih_f, EMB, 0, wih_f_bf, 4 * HH, EMB);
  cvt_tileB_k<<<cdiv(4 * HH * EMB, TB), TB, 0, stream>>>(enc_w_ih_r, EMB, 0, wih_r_bf, 4 * HH, EMB);
  cvt_tileB_k<<<cdiv(4 * HH * HH, TB), TB, 0, stream>>>(enc_w_hh_f, HH, 0, whh_f_bf, 4 * HH, HH);
  cvt_tileB_k<<<cdiv(4 * HH * HH, TB), TB, 0, stream>>>(enc_w_hh_r, HH, 0, whh_r_bf, 4 * HH, HH);
  cvt_tileB_k<<<cdiv(4 * DH * EMB, TB), TB, 0, stream>>>(dec_w_ih, 1536, 0, wih_e_bf, 4 * DH, EMB);
  cvt_tile_wcat_k<<<cdiv(4 * DH * 2048, TB), TB, 0, stream>>>(dec_w_ih, dec_w_hh, wcat_bf);
  cvt_tileB_k<<<cdiv(DH * DH, TB), TB, 0, stream>>>(attn_in, DH, 0, ain_bf, DH, DH);
  cvt_tileB_k<<<cdiv(DH * 2048, TB), TB, 0, stream>>>(attn_out, 2048, 0, aout_bf, DH, 2048);
  cvt_tileB_k<<<cdiv(VOCAB * DH, TB), TB, 0, stream>>>(gen_w, DH, 0, genw_bf, VOCAB, DH);

  // 3. hoisted input projections (bias folded in)
  {
    int mt = (S_LEN * BATCH) / 16, ng = (4 * HH) / 64;  // 128, 32
    int blocks = cdiv(mt * ng * 32, TB);
    gemm_bf16_k<<<blocks, TB, 0, stream>>>(xsrc_bf, EMB, wih_f_bf, enc_b_f, PF, 4 * HH, mt, ng, EMB);
    gemm_bf16_k<<<blocks, TB, 0, stream>>>(xsrc_bf, EMB, wih_r_bf, enc_b_r, PR, 4 * HH, mt, ng, EMB);
  }
  {
    int mt = DROWS / 16, ng = (4 * DH) / 64;  // 98, 64
    gemm_bf16_k<<<cdiv(mt * ng * 32, TB), TB, 0, stream>>>(xtrg_bf, EMB, wih_e_bf, dec_b, PD, 4 * DH, mt, ng, EMB);
  }

  // 4. persistent encoder (dir 0 = fwd, dir 1 = rev), 16 waves each
  {
    size_t lds = (size_t)2 * BATCH * HH * sizeof(bf16) + (size_t)BATCH * HH * sizeof(float);
    encoder_k<<<2, 512, lds, stream>>>(PF, PR, whh_f_bf, whh_r_bf, ctx, hfin, cfin);
  }

  // 5. persistent decoder + attention, 32 waves
  {
    size_t lds = (size_t)BATCH * 2048 * 2 + 2 * (size_t)BATCH * DH * 2 + (size_t)BATCH * S_LEN * 4;
    decoder_k<<<1, 1024, lds, stream>>>(PD, wcat_bf, ain_bf, aout_bf, ctx, src_mask,
                                        hfin, hfin + BATCH * HH, cfin, cfin + BATCH * HH,
                                        c_glob, dec_out);
  }

  // 6. generator logits -> d_out
  {
    int mt = DROWS / 16, ng = VOCAB / 64;  // 98, 500
    gemm_bf16_k<<<cdiv(mt * ng * 32, TB), TB, 0, stream>>>(dec_out, DH, genw_bf, gen_b, out, VOCAB, mt, ng, DH);
  }

  // 7. in-place log_softmax
  logsoftmax_k<<<DROWS, 256, 0, stream>>>(out);
}
